// VoxelFusion_36919538876794
// MI455X (gfx1250) — matrix operand order repaired and checked by reading, not compiled, not run on hardware
//
#include <hip/hip_runtime.h>
#include <hip/hip_bf16.h>

typedef __bf16 v16bf __attribute__((ext_vector_type(16)));
typedef __bf16 v8bf  __attribute__((ext_vector_type(8)));
typedef float  v8f   __attribute__((ext_vector_type(8)));

#define NVOX  24000
#define TILE  16
#define NOFF  11
#define PAIRS (TILE*NOFF)   // 176
#define CC    256
#define H8    116
#define W8    200
#define HW    (H8*W8)       // 23200

// workspace offsets (bf16 elements); all weights stored [col][k] contiguous-in-k
#define OFF_WQT 0           // 256 cols x 128 k
#define OFF_WK  32768       // 256 cols x 256 k   (w_k row-major == [c][n] already)
#define OFF_WVT 98304       // 256 cols x 256 k
#define OFF_WCT 163840      // 192 cols x 256 k
#define OFF_WFT 212992      // 128 cols x 320 k
#define WS_ELEMS 253952     // 507904 bytes required in d_ws

// LDS layout (bytes)
#define L_A    0            // cam_feat bf16 [176][256] = 90112
#define L_Q    90112        // f32 [16][256] : Q, later qk = Q @ Wk^T
#define L_QH   106496       // bf16 [16][256] Q
#define L_X    114688       // bf16 [16][128] voxel features
#define L_AA   118784       // bf16 [16][256] a = A^T sim
#define L_R    126976       // bf16 [16][256] res
#define L_CAT  135168       // bf16 [16][320] concat(cross, x)
#define L_SIM  145408       // f32 [176]
#define L_BKQ  146432       // f32 [16]
#define L_PB   146688       // int [176] gather base
#define L_PO   147712       // int [176][4] tap offsets
#define L_PW   150784       // f32 [176][4] tap weights
#define SMEM_BYTES 153856

__global__ void vx_transpose_bf16(const float* __restrict__ src, __bf16* __restrict__ dst,
                                  int K, int Nn) {
  int i = blockIdx.x * 256 + threadIdx.x;           // i = n*K + k
  if (i < K * Nn) {
    int n = i / K;
    int k = i - n * K;
    dst[i] = (__bf16)src[(size_t)k * Nn + n];
  }
}

__global__ void vx_copy_bf16(const float* __restrict__ src, __bf16* __restrict__ dst, int n) {
  int i = blockIdx.x * 256 + threadIdx.x;
  if (i < n) dst[i] = (__bf16)src[i];
}

// One 16x16 output tile: A is 16 x (32*KSTEPS) row-major (stride sA, bf16, LDS),
// B column `col` is contiguous-in-k at Bw + col*sB (bf16, global).
// A fragment per ISA: lane holds row (lane&15); K groups {0..7,16..23} for lanes 0-15,
// {8..15,24..31} for lanes 16-31. B fragment: lane holds column (lane&15), same K groups as A.
template <int KSTEPS>
__device__ inline v8f gemm_row(const __bf16* __restrict__ A, int sA,
                               const __bf16* __restrict__ Bw, int sB,
                               int col, int lane) {
  v8f acc = {};
  const __bf16* arow = A + (lane & 15) * sA + ((lane >> 4) << 3);
  const __bf16* bcol = Bw + (size_t)col * sB + ((lane >> 4) << 3);
#pragma unroll
  for (int ks = 0; ks < KSTEPS; ++ks) {
    const __bf16* ap = arow + ks * 32;
    v8bf lo = *(const v8bf*)(ap);
    v8bf hi = *(const v8bf*)(ap + 16);
    v16bf a;
#pragma unroll
    for (int e = 0; e < 8; ++e) { a[e] = lo[e]; a[e + 8] = hi[e]; }
    v8bf blo = *(const v8bf*)(bcol + ks * 32);
    v8bf bhi = *(const v8bf*)(bcol + ks * 32 + 16);
    v16bf b;
#pragma unroll
    for (int e = 0; e < 8; ++e) { b[e] = blo[e]; b[e + 8] = bhi[e]; }
    acc = __builtin_amdgcn_wmma_f32_16x16x32_bf16(false, a, false, b, (short)0, acc,
                                                  false, false);
  }
  return acc;
}

__global__ __launch_bounds__(256) void vx_fused(
    const float* __restrict__ vf, const int* __restrict__ vc,
    const float* __restrict__ vm, const float* __restrict__ img,
    const float* __restrict__ Tm, const float* __restrict__ Km,
    const __bf16* __restrict__ wsw,
    const float* __restrict__ bq, const float* __restrict__ bk,
    const float* __restrict__ bv, const float* __restrict__ bc,
    const float* __restrict__ bfw, float* __restrict__ out) {
  extern __shared__ __align__(256) char smem[];
  __bf16* smA   = (__bf16*)(smem + L_A);
  float*  smQ   = (float*) (smem + L_Q);
  __bf16* smQh  = (__bf16*)(smem + L_QH);
  __bf16* smX   = (__bf16*)(smem + L_X);
  __bf16* smAa  = (__bf16*)(smem + L_AA);
  __bf16* smR   = (__bf16*)(smem + L_R);
  __bf16* smCat = (__bf16*)(smem + L_CAT);
  float*  smSim = (float*) (smem + L_SIM);
  float*  smBkq = (float*) (smem + L_BKQ);
  int*    pB    = (int*)   (smem + L_PB);
  int*    pO    = (int*)   (smem + L_PO);
  float*  pW    = (float*) (smem + L_PW);

  const int tid  = threadIdx.x;
  const int wave = tid >> 5;
  const int lane = tid & 31;
  const int blk  = blockIdx.x;

  // hint the L2-resident weights toward the cache hierarchy (global_prefetch_b8)
  __builtin_prefetch(wsw + (tid << 6), 0, 1);

  // ---- Phase A: projection params (1 thread per (voxel, offset)) + stage X as bf16 ----
  if (tid < PAIRS) {
    int m = tid / NOFF, j = tid - m * NOFF;
    int g = blk * TILE + m;
    float mx = vm[g * 4 + 0], my = vm[g * 4 + 1], mz = vm[g * 4 + 2];
    int cam = (int)roundf(vm[g * 4 + 3]);
    int bid = vc[g * 4 + 0];
    const float* T = Tm + cam * 16;
    float pc0 = T[0] * mx + T[1] * my + T[2]  * mz + T[3];
    float pc1 = T[4] * mx + T[5] * my + T[6]  * mz + T[7];
    float pc2 = T[8] * mx + T[9] * my + T[10] * mz + T[11];
    const float* Kc = Km + cam * 9;
    float q0 = Kc[0] * 0.125f * pc0 + Kc[1] * pc1 + Kc[2] * 0.125f * pc2;
    float q1 = Kc[3] * pc0 + Kc[4] * 0.125f * pc1 + Kc[5] * 0.125f * pc2;
    float q2 = Kc[6] * pc0 + Kc[7] * pc1 + Kc[8] * pc2;
    float u = q0 / q2, v = q1 / q2;
    // grid-sample algebra in the reference collapses to ix = clip(u+dx,0,200)-0.5 etc.
    float dx = -1.0f + 0.2f * (float)j;
    float ix = fminf(fmaxf(u + dx, 0.f), 200.f) - 0.5f;
    float iy = fminf(fmaxf(v, 0.f), 200.f) - 0.5f;
    float x0f = floorf(ix), y0f = floorf(iy);
    float wx = ix - x0f, wy = iy - y0f;
    int x0 = (int)x0f, y0 = (int)y0f;
    pB[tid] = ((cam * 2 + bid) * CC) * HW;
    float tws[4] = {(1.f - wy) * (1.f - wx), (1.f - wy) * wx, wy * (1.f - wx), wy * wx};
    int xs[4] = {x0, x0 + 1, x0, x0 + 1};
    int ys[4] = {y0, y0, y0 + 1, y0 + 1};
#pragma unroll
    for (int kk = 0; kk < 4; ++kk) {
      bool valid = (xs[kk] >= 0) && (xs[kk] <= W8 - 1) && (ys[kk] >= 0) && (ys[kk] <= H8 - 1);
      int xi = min(max(xs[kk], 0), W8 - 1);
      int yi = min(max(ys[kk], 0), H8 - 1);
      pO[tid * 4 + kk] = yi * W8 + xi;
      pW[tid * 4 + kk] = valid ? tws[kk] : 0.f;
    }
  }
  for (int i = tid; i < TILE * 128; i += 256) {
    int m = i >> 7, k = i & 127;
    smX[i] = (__bf16)vf[(size_t)(blk * TILE + m) * 128 + k];
  }
  __syncthreads();

  // ---- Phase B: bilinear gather cam_feat -> smA bf16 [176][256] ----
  // pair index fastest so adjacent lanes hit adjacent x (coalesced within a voxel)
  for (int i = tid; i < PAIRS * CC; i += 256) {
    int c = i / PAIRS;
    int p = i - c * PAIRS;
    const float* ib = img + pB[p] + c * HW;
    float val = pW[p * 4 + 0] * ib[pO[p * 4 + 0]] + pW[p * 4 + 1] * ib[pO[p * 4 + 1]] +
                pW[p * 4 + 2] * ib[pO[p * 4 + 2]] + pW[p * 4 + 3] * ib[pO[p * 4 + 3]];
    smA[p * CC + c] = (__bf16)val;
  }
  __syncthreads();

  // ---- Phase C1: Q = X @ Wq + bq   (M=16, K=128, N=256; 2 n-tiles per wave) ----
#pragma unroll
  for (int rep = 0; rep < 2; ++rep) {
    int nt = wave * 2 + rep;
    int col = nt * 16 + (lane & 15);
    v8f acc = gemm_row<4>(smX, 128, wsw + OFF_WQT, 128, col, lane);
    float bb = bq[col];
#pragma unroll
    for (int r = 0; r < 8; ++r) {
      int row = r + ((lane >> 4) << 3);
      float qv = acc[r] + bb;
      smQ[row * CC + col]  = qv;
      smQh[row * CC + col] = (__bf16)qv;
    }
  }
  __syncthreads();

  // ---- Phase C2: bkq[m] = b_k . Q[m]  ----
  if (tid < TILE) {
    float s = 0.f;
    for (int n = 0; n < CC; ++n) s += bk[n] * smQ[tid * CC + n];
    smBkq[tid] = s;
  }
  __syncthreads();

  // ---- Phase C3: qk = Q @ Wk^T  (overwrite smQ with qk, f32) ----
#pragma unroll
  for (int rep = 0; rep < 2; ++rep) {
    int nt = wave * 2 + rep;
    int col = nt * 16 + (lane & 15);                 // col = input-channel c of w_k
    v8f acc = gemm_row<8>(smQh, CC, wsw + OFF_WK, CC, col, lane);
#pragma unroll
    for (int r = 0; r < 8; ++r) {
      int row = r + ((lane >> 4) << 3);
      smQ[row * CC + col] = acc[r];
    }
  }
  __syncthreads();

  // ---- Phase D1: sim_raw[p] = A[p,:] . qk[m,:] + bkq[m]  (wave per pair, lane-split dot) ----
  for (int p = wave; p < PAIRS; p += 8) {
    int m = p / NOFF;
    float s = 0.f;
    for (int c = lane; c < CC; c += 32) s += (float)smA[p * CC + c] * smQ[m * CC + c];
    s += __shfl_xor(s, 16, 32);
    s += __shfl_xor(s, 8, 32);
    s += __shfl_xor(s, 4, 32);
    s += __shfl_xor(s, 2, 32);
    s += __shfl_xor(s, 1, 32);
    if (lane == 0) smSim[p] = s + smBkq[m];
  }
  __syncthreads();

  // ---- Phase D2: softmax over 11 offsets per voxel ----
  if (tid < TILE) {
    float mx = -3.4e38f;
    for (int j = 0; j < NOFF; ++j) mx = fmaxf(mx, smSim[tid * NOFF + j]);
    float sum = 0.f;
    float e[NOFF];
    for (int j = 0; j < NOFF; ++j) { e[j] = __expf(smSim[tid * NOFF + j] - mx); sum += e[j]; }
    float inv = 1.f / sum;
    for (int j = 0; j < NOFF; ++j) smSim[tid * NOFF + j] = e[j] * inv;
  }
  __syncthreads();

  // ---- Phase D3: a[m][c] = sum_j sim * A  -> bf16 ----
  for (int i = tid; i < TILE * CC; i += 256) {
    int m = i >> 8, c = i & 255;
    float s = 0.f;
#pragma unroll
    for (int j = 0; j < NOFF; ++j)
      s += smSim[m * NOFF + j] * (float)smA[(m * NOFF + j) * CC + c];
    smAa[m * CC + c] = (__bf16)s;
  }
  __syncthreads();

  // ---- Phase E: res = a @ Wv + bv  (softmax sums to 1) ----
#pragma unroll
  for (int rep = 0; rep < 2; ++rep) {
    int nt = wave * 2 + rep;
    int col = nt * 16 + (lane & 15);
    v8f acc = gemm_row<8>(smAa, CC, wsw + OFF_WVT, CC, col, lane);
    float bb = bv[col];
#pragma unroll
    for (int r = 0; r < 8; ++r) {
      int row = r + ((lane >> 4) << 3);
      smR[row * CC + col] = (__bf16)(acc[r] + bb);
    }
  }
  __syncthreads();

  // ---- Phase F: cross = res @ Wc + bc (N=192), and stage concat[.,192:320] = X ----
  for (int i = tid; i < TILE * 128; i += 256) {
    int m = i >> 7, k = i & 127;
    smCat[m * 320 + 192 + k] = smX[i];
  }
  for (int rep = 0; rep < 2; ++rep) {
    int nt = wave + rep * 8;
    if (nt >= 12) break;                               // wave-uniform
    int col = nt * 16 + (lane & 15);
    v8f acc = gemm_row<8>(smR, CC, wsw + OFF_WCT, CC, col, lane);
    float bb = bc[col];
#pragma unroll
    for (int r = 0; r < 8; ++r) {
      int row = r + ((lane >> 4) << 3);
      smCat[row * 320 + col] = (__bf16)(acc[r] + bb);
    }
  }
  __syncthreads();

  // ---- Phase G: out = concat @ Wf + bf  (M=16, K=320, N=128; 1 n-tile per wave) ----
  {
    int col = wave * 16 + (lane & 15);
    v8f acc = gemm_row<10>(smCat, 320, wsw + OFF_WFT, 320, col, lane);
    float bb = bfw[col];
#pragma unroll
    for (int r = 0; r < 8; ++r) {
      int row = r + ((lane >> 4) << 3);
      out[(size_t)(blk * TILE + row) * 128 + col] = acc[r] + bb;
    }
  }
}

extern "C" void kernel_launch(void* const* d_in, const int* in_sizes, int n_in,
                              void* d_out, int out_size, void* d_ws, size_t ws_size,
                              hipStream_t stream) {
  const float* vf  = (const float*)d_in[0];
  const int*   vc  = (const int*)  d_in[1];
  const float* vm  = (const float*)d_in[2];
  const float* img = (const float*)d_in[3];
  const float* Tm  = (const float*)d_in[4];
  const float* Km  = (const float*)d_in[5];
  const float* wq  = (const float*)d_in[6];
  const float* bq  = (const float*)d_in[7];
  const float* wk  = (const float*)d_in[8];
  const float* bk  = (const float*)d_in[9];
  const float* wv  = (const float*)d_in[10];
  const float* bv  = (const float*)d_in[11];
  const float* wc  = (const float*)d_in[12];
  const float* bc  = (const float*)d_in[13];
  const float* wf  = (const float*)d_in[14];
  const float* bf  = (const float*)d_in[15];
  __bf16* ws = (__bf16*)d_ws;

  // Prologue: bf16 weights in [col][k] layout (L2-resident, shared by all blocks)
  vx_transpose_bf16<<<(32768 + 255) / 256, 256, 0, stream>>>(wq, ws + OFF_WQT, 128, 256);
  vx_copy_bf16     <<<(65536 + 255) / 256, 256, 0, stream>>>(wk, ws + OFF_WK, 65536);
  vx_transpose_bf16<<<(65536 + 255) / 256, 256, 0, stream>>>(wv, ws + OFF_WVT, 256, 256);
  vx_transpose_bf16<<<(49152 + 255) / 256, 256, 0, stream>>>(wc, ws + OFF_WCT, 256, 192);
  vx_transpose_bf16<<<(40960 + 255) / 256, 256, 0, stream>>>(wf, ws + OFF_WFT, 320, 128);

  vx_fused<<<NVOX / TILE, 256, SMEM_BYTES, stream>>>(vf, vc, vm, img, Tm, Km, ws,
                                                     bq, bk, bv, bc, bf, (float*)d_out);
}
